// NonLocalBlock1D_multi_head_new_5257039970803
// MI455X (gfx1250) — matrix operand order repaired and checked by reading, not compiled, not run on hardware
//
#include <hip/hip_runtime.h>

#define B_ 16
#define C_ 512
#define T_ 1024
#define INTER_ 256
#define OUT_ 512
#define H_ 3
#define TL_ 2
#define EPS_ 1e-5f

typedef __bf16 bf16_t;
typedef __attribute__((ext_vector_type(16))) __bf16 v16bf;
typedef __attribute__((ext_vector_type(8)))  __bf16 v8bf;
typedef __attribute__((ext_vector_type(8)))  float  v8f;

__device__ __forceinline__ v8f wmma_bf16(v16bf a, v16bf b, v8f c) {
  // D = A(16x32) * B(32x16) + C, fp32 accumulate
  return __builtin_amdgcn_wmma_f32_16x16x32_bf16(false, a, false, b, (short)0, c, false, false);
}

// A fragment (16x32 bf16, row-major source, stride lda):
// lane L: row m = m_base + (L&15); K-base = k_base + (L>>4)*16, halves at +0 and +8.
__device__ __forceinline__ v16bf load_a_frag(const bf16_t* A, int m_base, int lda, int k_base) {
  const int lane = threadIdx.x & 31;
  const int m  = m_base + (lane & 15);
  const int kb = k_base + ((lane >> 4) << 4);
  const bf16_t* p = A + (size_t)m * lda + kb;
  union { v16bf v; v8bf h[2]; } u;
  u.h[0] = *(const v8bf*)(p);
  u.h[1] = *(const v8bf*)(p + 8);
  return u.v;
}

// B fragment (32x16 bf16) loaded from Bt = B^T (N x K row-major, stride ldb):
// lane L: col n = n_base + (L&15); K-base = k_base + (L>>4)*16, 16 contiguous.
__device__ __forceinline__ v16bf load_b_frag(const bf16_t* Bt, int n_base, int ldb, int k_base) {
  const int lane = threadIdx.x & 31;
  const int n  = n_base + (lane & 15);
  const int kb = k_base + ((lane >> 4) << 4);
  return *(const v16bf*)(Bt + (size_t)n * ldb + kb);
}

// ------- double-buffered GEMM mainloop: wave tile 32(M) x 64(N), K templated -------
struct Frags { v16bf a[2]; v16bf b[4]; };

__device__ __forceinline__ Frags load_frags(const bf16_t* A, int m_wave, int lda,
                                            const bf16_t* Bt, int n_base, int ldb, int kc) {
  Frags f;
  f.a[0] = load_a_frag(A, m_wave,      lda, kc);
  f.a[1] = load_a_frag(A, m_wave + 16, lda, kc);
#pragma unroll
  for (int jn = 0; jn < 4; ++jn) f.b[jn] = load_b_frag(Bt, n_base + jn * 16, ldb, kc);
  return f;
}

template <int K>
__device__ __forceinline__ void gemm_mainloop(const bf16_t* A, int m_wave, int lda,
                                              const bf16_t* Bt, int n_base, int ldb,
                                              v8f acc[2][4]) {
  Frags cur = load_frags(A, m_wave, lda, Bt, n_base, ldb, 0);
#pragma unroll
  for (int kc = 0; kc < K; kc += 32) {
    Frags nxt = cur;
    if (kc + 32 < K) nxt = load_frags(A, m_wave, lda, Bt, n_base, ldb, kc + 32); // prefetch
#pragma unroll
    for (int mi = 0; mi < 2; ++mi)
#pragma unroll
      for (int jn = 0; jn < 4; ++jn)
        acc[mi][jn] = wmma_bf16(cur.a[mi], cur.b[jn], acc[mi][jn]);
    cur = nxt;
  }
}

// ---------------- weight conversion ----------------
__global__ void k_cvt_f32_bf16(const float* __restrict__ src, bf16_t* __restrict__ dst, int n) {
  int i = blockIdx.x * blockDim.x + threadIdx.x;
  if (i < n) dst[i] = (bf16_t)src[i];
}

__global__ void k_cvt_proj(const float* __restrict__ g_w, const float* __restrict__ th_w,
                           const float* __restrict__ ph_w, const float* __restrict__ g_b,
                           const float* __restrict__ th_b, const float* __restrict__ ph_b,
                           bf16_t* __restrict__ projW, float* __restrict__ projB) {
  int i = blockIdx.x * blockDim.x + threadIdx.x;
  const int total = H_ * 768 * C_;
  if (i < total) {
    int c = i % C_;
    int o = (i / C_) % 768;
    int h = i / (C_ * 768);
    float v;
    if (o < 256)      v = g_w [(h * 256 + o)       * C_ + c];
    else if (o < 512) v = th_w[(h * 256 + o - 256) * C_ + c];
    else              v = ph_w[(h * 256 + o - 512) * C_ + c];
    projW[i] = (bf16_t)v;
  }
  if (i < H_ * 768) {
    int o = i % 768, h = i / 768;
    float v;
    if (o < 256)      v = g_b [h * 256 + o];
    else if (o < 512) v = th_b[h * 256 + o - 256];
    else              v = ph_b[h * 256 + o - 512];
    projB[i] = v;
  }
}

// ---------------- x (B,C,T) f32 -> xT (B,T,C) bf16 ----------------
__global__ void k_transpose_x(const float* __restrict__ x, bf16_t* __restrict__ xT) {
  __shared__ float tile[32][33];
  const int b  = blockIdx.z;
  const int t0 = blockIdx.x * 32, c0 = blockIdx.y * 32;
  const int tx = threadIdx.x, ty = threadIdx.y;
#pragma unroll
  for (int r = 0; r < 32; r += 8)
    tile[ty + r][tx] = x[((size_t)b * C_ + (c0 + ty + r)) * T_ + (t0 + tx)];
  __syncthreads();
#pragma unroll
  for (int r = 0; r < 32; r += 8)
    xT[((size_t)b * T_ + (t0 + ty + r)) * C_ + (c0 + tx)] = (bf16_t)tile[tx][ty + r];
}

// ---------------- temporal dilated conv as GEMM -> featsT[layer] ----------------
// feats[o,t] = sum_{kh,c} w[o,kh,c] * x[c, t+(kh-1)*d]; stored transposed (t,c) bf16.
__global__ __launch_bounds__(128) void k_tconv(const bf16_t* __restrict__ convW,
                                               const bf16_t* __restrict__ xT,
                                               bf16_t* __restrict__ featsT) {
  const int lane = threadIdx.x & 31, wid = threadIdx.x >> 5;
  const int col = lane & 15, half = lane >> 4;
  const int z = blockIdx.z;
  const int layer = z / B_, b = z % B_;
  const int d = layer + 1;
  const int m_wave = blockIdx.y * 128 + wid * 32;
  const int n_base = blockIdx.x * 64;
  const bf16_t* A   = convW + (size_t)layer * C_ * 3 * C_;
  const bf16_t* xTb = xT + (size_t)b * T_ * C_;
  v8f acc[2][4] = {};
  const bool edge = (blockIdx.x == 0) || (blockIdx.x == gridDim.x - 1);
  if (!edge) {
    // interior: shifted-by-off rows all in range; just offset the B^T base pointer
#pragma unroll
    for (int kh = 0; kh < 3; ++kh) {
      const int off = (kh - 1) * d;
      gemm_mainloop<C_>(A + kh * C_, m_wave, 3 * C_, xTb + (ptrdiff_t)off * C_, n_base, C_, acc);
    }
  } else {
    const v16bf zv = {};
#pragma unroll
    for (int kh = 0; kh < 3; ++kh) {
      const int off = (kh - 1) * d;
      for (int kc = 0; kc < C_; kc += 32) {
        v16bf a0 = load_a_frag(A + kh * C_, m_wave,      3 * C_, kc);
        v16bf a1 = load_a_frag(A + kh * C_, m_wave + 16, 3 * C_, kc);
#pragma unroll
        for (int jn = 0; jn < 4; ++jn) {
          int t = n_base + jn * 16 + col + off;
          bool ok = (unsigned)t < (unsigned)T_;
          int tc = ok ? t : 0;
          int kb = kc + (half << 4);
          v16bf bv = *(const v16bf*)(xTb + (size_t)tc * C_ + kb);
          bv = ok ? bv : zv;               // zero-pad shifted-out rows
          acc[0][jn] = wmma_bf16(a0, bv, acc[0][jn]);
          acc[1][jn] = wmma_bf16(a1, bv, acc[1][jn]);
        }
      }
    }
  }
  bf16_t* out = featsT + (size_t)(layer * B_ + b) * T_ * C_;
#pragma unroll
  for (int mi = 0; mi < 2; ++mi) {
    const int c0 = m_wave + mi * 16 + half * 8;
#pragma unroll
    for (int jn = 0; jn < 4; ++jn) {
      int t = n_base + jn * 16 + col;
      v8bf pk;
#pragma unroll
      for (int j = 0; j < 8; ++j) pk[j] = (bf16_t)acc[mi][jn][j];
      *(v8bf*)(out + (size_t)t * C_ + c0) = pk;
    }
  }
}

// ---------------- per-head g/theta/phi projection GEMM ----------------
__global__ __launch_bounds__(128) void k_proj(const bf16_t* __restrict__ projW,
                                              const float* __restrict__ projB,
                                              const bf16_t* __restrict__ featsT,
                                              bf16_t* __restrict__ gbuf,
                                              bf16_t* __restrict__ thT,
                                              bf16_t* __restrict__ phT) {
  const int lane = threadIdx.x & 31, wid = threadIdx.x >> 5;
  const int col = lane & 15, half = lane >> 4;
  const int hb = blockIdx.z, h = hb / B_;
  const int m_wave = blockIdx.y * 128 + wid * 32;
  const int n_base = blockIdx.x * 64;
  const bf16_t* A  = projW + (size_t)h * 768 * C_;
  const bf16_t* Bt = featsT + (size_t)hb * T_ * C_;
  v8f acc[2][4] = {};
  gemm_mainloop<C_>(A, m_wave, C_, Bt, n_base, C_, acc);

  const int seg = m_wave >> 8;          // 0:g, 1:theta, 2:phi (32-row wave tile stays in one seg)
  const float* bias = projB + h * 768;
#pragma unroll
  for (int mi = 0; mi < 2; ++mi) {
    const int m0 = m_wave + mi * 16 + half * 8;
    if (seg == 0) {                     // g stored natural (i-major over t)
#pragma unroll
      for (int jn = 0; jn < 4; ++jn) {
        int t = n_base + jn * 16 + col;
#pragma unroll
        for (int j = 0; j < 8; ++j)
          gbuf[((size_t)hb * INTER_ + (m0 + j)) * T_ + t] = (bf16_t)(acc[mi][jn][j] + bias[m0 + j]);
      }
    } else {                            // theta/phi stored transposed (t, i)
      bf16_t* dst = (seg == 1 ? thT : phT);
      const int ioff = m0 - seg * 256;
#pragma unroll
      for (int jn = 0; jn < 4; ++jn) {
        int t = n_base + jn * 16 + col;
        v8bf pk;
#pragma unroll
        for (int j = 0; j < 8; ++j) pk[j] = (bf16_t)(acc[mi][jn][j] + bias[m0 + j]);
        *(v8bf*)(dst + ((size_t)hb * T_ + t) * INTER_ + ioff) = pk;
      }
    }
  }
}

// ---------------- flash attention: one wave = 16 query rows, full 256 out-channels ----------------
__global__ __launch_bounds__(128) void k_attn(const bf16_t* __restrict__ thT,
                                              const bf16_t* __restrict__ phT,
                                              const bf16_t* __restrict__ gbuf,
                                              bf16_t* __restrict__ yT) {
  __shared__ __align__(32) bf16_t lds_p[4][16][32];
  const int lane = threadIdx.x & 31, wid = threadIdx.x >> 5;
  const int col = lane & 15, half = lane >> 4;
  const int hb = blockIdx.y, h = hb / B_, b = hb % B_;
  const int qbase = (blockIdx.x * 4 + wid) * 16;
  const bf16_t* thx = thT + (size_t)hb * T_ * INTER_;
  const bf16_t* phx = phT + (size_t)hb * T_ * INTER_;
  const bf16_t* gx  = gbuf + (size_t)hb * INTER_ * T_;
  v8f acc[16] = {};
  float mrun[8], lrun[8];
#pragma unroll
  for (int j = 0; j < 8; ++j) { mrun[j] = -1e30f; lrun[j] = 0.f; }

  for (int sb = 0; sb < T_; sb += 32) {
    v8f s0 = {}, s1 = {};
    // pipelined score K-loop (K = INTER = 256)
    v16bf a  = load_a_frag(thx, qbase,   INTER_, 0);
    v16bf b0 = load_b_frag(phx, sb,      INTER_, 0);
    v16bf b1 = load_b_frag(phx, sb + 16, INTER_, 0);
#pragma unroll
    for (int ks = 0; ks < 8; ++ks) {
      v16bf ca = a, cb0 = b0, cb1 = b1;
      if (ks < 7) {
        a  = load_a_frag(thx, qbase,   INTER_, (ks + 1) * 32);
        b0 = load_b_frag(phx, sb,      INTER_, (ks + 1) * 32);
        b1 = load_b_frag(phx, sb + 16, INTER_, (ks + 1) * 32);
      }
      s0 = wmma_bf16(ca, cb0, s0);
      s1 = wmma_bf16(ca, cb1, s1);
    }
    // online softmax per row (row m = j + 8*half lives across 16 lanes of a half-wave)
    float scl[8];
#pragma unroll
    for (int j = 0; j < 8; ++j) {
      float mx = fmaxf(s0[j], s1[j]);
      mx = fmaxf(mx, __shfl_xor(mx, 1, 32));
      mx = fmaxf(mx, __shfl_xor(mx, 2, 32));
      mx = fmaxf(mx, __shfl_xor(mx, 4, 32));
      mx = fmaxf(mx, __shfl_xor(mx, 8, 32));
      float mnew = fmaxf(mrun[j], mx);
      scl[j] = __expf(mrun[j] - mnew);
      float p0 = __expf(s0[j] - mnew);
      float p1 = __expf(s1[j] - mnew);
      s0[j] = p0; s1[j] = p1;
      float rs = p0 + p1;
      rs += __shfl_xor(rs, 1, 32);
      rs += __shfl_xor(rs, 2, 32);
      rs += __shfl_xor(rs, 4, 32);
      rs += __shfl_xor(rs, 8, 32);
      lrun[j] = lrun[j] * scl[j] + rs;
      mrun[j] = mnew;
    }
#pragma unroll
    for (int it = 0; it < 16; ++it)
#pragma unroll
      for (int j = 0; j < 8; ++j) acc[it][j] *= scl[j];
    // re-layout p: D-fragment -> A-fragment via LDS
#pragma unroll
    for (int j = 0; j < 8; ++j) {
      lds_p[wid][j + 8 * half][col]      = (bf16_t)s0[j];
      lds_p[wid][j + 8 * half][16 + col] = (bf16_t)s1[j];
    }
    asm volatile("s_wait_dscnt 0" ::: "memory");
    union { v16bf v; v8bf hh[2]; } up;
    up.hh[0] = *(const v8bf*)&lds_p[wid][col][half * 16];
    up.hh[1] = *(const v8bf*)&lds_p[wid][col][half * 16 + 8];
    // pipelined y-update: y[t,i] += p[t,s] * g[i,s]
    v16bf bg = load_b_frag(gx, 0, T_, sb);
#pragma unroll
    for (int it = 0; it < 16; ++it) {
      v16bf cb = bg;
      if (it < 15) bg = load_b_frag(gx, (it + 1) * 16, T_, sb);
      acc[it] = wmma_bf16(up.v, cb, acc[it]);
    }
  }
  float inv[8];
#pragma unroll
  for (int j = 0; j < 8; ++j) inv[j] = 1.0f / lrun[j];
#pragma unroll
  for (int it = 0; it < 16; ++it)
#pragma unroll
    for (int j = 0; j < 8; ++j) {
      int t = qbase + j + 8 * half;
      int i = h * 256 + it * 16 + col;
      yT[((size_t)b * T_ + t) * 768 + i] = (bf16_t)(acc[it][j] * inv[j]);
    }
}

// ---------------- W conv + BN1 + residual -> zT ----------------
__global__ __launch_bounds__(128) void k_wconv(const bf16_t* __restrict__ Wb,
                                               const float* __restrict__ W_bias,
                                               const float* __restrict__ g1, const float* __restrict__ be1,
                                               const float* __restrict__ mu1, const float* __restrict__ va1,
                                               const bf16_t* __restrict__ yT,
                                               const float* __restrict__ x,
                                               bf16_t* __restrict__ zT) {
  const int lane = threadIdx.x & 31, wid = threadIdx.x >> 5;
  const int col = lane & 15, half = lane >> 4;
  const int b = blockIdx.z;
  const int m_wave = blockIdx.y * 128 + wid * 32;
  const int n_base = blockIdx.x * 64;
  const bf16_t* Bt = yT + (size_t)b * T_ * 768;
  v8f acc[2][4] = {};
  gemm_mainloop<768>(Wb, m_wave, 768, Bt, n_base, 768, acc);
#pragma unroll
  for (int mi = 0; mi < 2; ++mi) {
    const int m0 = m_wave + mi * 16 + half * 8;
    float bias[8], sc[8], sh[8];
#pragma unroll
    for (int j = 0; j < 8; ++j) {
      int m = m0 + j;
      float s = g1[m] * rsqrtf(va1[m] + EPS_);
      sc[j] = s; sh[j] = be1[m] - mu1[m] * s; bias[j] = W_bias[m];
    }
#pragma unroll
    for (int jn = 0; jn < 4; ++jn) {
      int t = n_base + jn * 16 + col;
      v8bf pk;
#pragma unroll
      for (int j = 0; j < 8; ++j) {
        float v = (acc[mi][jn][j] + bias[j]) * sc[j] + sh[j];
        v += x[((size_t)b * C_ + (m0 + j)) * T_ + t];   // residual
        pk[j] = (bf16_t)v;
      }
      *(v8bf*)(zT + ((size_t)b * T_ + t) * C_ + m0) = pk;
    }
  }
}

// ---------------- f_x conv + BN2 -> out (f32) ----------------
__global__ __launch_bounds__(128) void k_fx(const bf16_t* __restrict__ fxb,
                                            const float* __restrict__ fx_bias,
                                            const float* __restrict__ g2, const float* __restrict__ be2,
                                            const float* __restrict__ mu2, const float* __restrict__ va2,
                                            const bf16_t* __restrict__ zT,
                                            float* __restrict__ out) {
  const int lane = threadIdx.x & 31, wid = threadIdx.x >> 5;
  const int col = lane & 15, half = lane >> 4;
  const int b = blockIdx.z;
  const int m_wave = blockIdx.y * 128 + wid * 32;
  const int n_base = blockIdx.x * 64;
  const bf16_t* Bt = zT + (size_t)b * T_ * C_;
  v8f acc[2][4] = {};
  gemm_mainloop<C_>(fxb, m_wave, C_, Bt, n_base, C_, acc);
#pragma unroll
  for (int mi = 0; mi < 2; ++mi) {
    const int m0 = m_wave + mi * 16 + half * 8;
    float bias[8], sc[8], sh[8];
#pragma unroll
    for (int j = 0; j < 8; ++j) {
      int m = m0 + j;
      float s = g2[m] * rsqrtf(va2[m] + EPS_);
      sc[j] = s; sh[j] = be2[m] - mu2[m] * s; bias[j] = fx_bias[m];
    }
#pragma unroll
    for (int jn = 0; jn < 4; ++jn) {
      int t = n_base + jn * 16 + col;
#pragma unroll
      for (int j = 0; j < 8; ++j)
        out[((size_t)b * OUT_ + (m0 + j)) * T_ + t] = (acc[mi][jn][j] + bias[j]) * sc[j] + sh[j];
    }
  }
}

extern "C" void kernel_launch(void* const* d_in, const int* in_sizes, int n_in,
                              void* d_out, int out_size, void* d_ws, size_t ws_size,
                              hipStream_t stream) {
  (void)in_sizes; (void)n_in; (void)out_size; (void)ws_size;
  const float* x        = (const float*)d_in[0];
  const float* tconv_w  = (const float*)d_in[1];
  const float* g_w      = (const float*)d_in[2];
  const float* g_b      = (const float*)d_in[3];
  const float* theta_w  = (const float*)d_in[4];
  const float* theta_b  = (const float*)d_in[5];
  const float* phi_w    = (const float*)d_in[6];
  const float* phi_b    = (const float*)d_in[7];
  const float* W_w      = (const float*)d_in[8];
  const float* W_b      = (const float*)d_in[9];
  const float* bn1g     = (const float*)d_in[10];
  const float* bn1b     = (const float*)d_in[11];
  const float* bn1m     = (const float*)d_in[12];
  const float* bn1v     = (const float*)d_in[13];
  const float* fx_w     = (const float*)d_in[14];
  const float* fx_b     = (const float*)d_in[15];
  const float* bn2g     = (const float*)d_in[16];
  const float* bn2b     = (const float*)d_in[17];
  const float* bn2m     = (const float*)d_in[18];
  const float* bn2v     = (const float*)d_in[19];

  char* w = (char*)d_ws;
  size_t off = 0;
  auto take = [&](size_t bytes) -> void* {
    void* p = w + off;
    off += (bytes + 255) & ~(size_t)255;
    return p;
  };
  bf16_t* featsT = (bf16_t*)take((size_t)H_ * B_ * T_ * C_ * 2);        // (h,b,t,c)
  bf16_t* convW  = (bf16_t*)take((size_t)TL_ * C_ * 3 * C_ * 2);
  bf16_t* projW  = (bf16_t*)take((size_t)H_ * 768 * C_ * 2);
  float*  projB  = (float*) take((size_t)H_ * 768 * 4);
  bf16_t* Wb16   = (bf16_t*)take((size_t)C_ * 768 * 2);
  bf16_t* fxb16  = (bf16_t*)take((size_t)OUT_ * C_ * 2);
  bf16_t* gbuf   = (bf16_t*)take((size_t)H_ * B_ * INTER_ * T_ * 2);    // (hb,i,t)
  bf16_t* thT    = (bf16_t*)take((size_t)H_ * B_ * T_ * INTER_ * 2);    // (hb,t,i)
  bf16_t* phT    = (bf16_t*)take((size_t)H_ * B_ * T_ * INTER_ * 2);    // (hb,t,i)
  bf16_t* yT     = (bf16_t*)take((size_t)B_ * T_ * 768 * 2);            // (b,t,h*256+i)
  bf16_t* zT     = (bf16_t*)take((size_t)B_ * T_ * C_ * 2);             // (b,t,c)

  // weight conversions
  {
    int n = TL_ * C_ * 3 * C_;
    k_cvt_f32_bf16<<<(n + 255) / 256, 256, 0, stream>>>(tconv_w, convW, n);
  }
  {
    int n = C_ * 768;
    k_cvt_f32_bf16<<<(n + 255) / 256, 256, 0, stream>>>(W_w, Wb16, n);
  }
  {
    int n = OUT_ * C_;
    k_cvt_f32_bf16<<<(n + 255) / 256, 256, 0, stream>>>(fx_w, fxb16, n);
  }
  {
    int n = H_ * 768 * C_;
    k_cvt_proj<<<(n + 255) / 256, 256, 0, stream>>>(g_w, theta_w, phi_w, g_b, theta_b, phi_b,
                                                    projW, projB);
  }
  // identity head features = xT
  bf16_t* xTh2 = featsT + (size_t)2 * B_ * T_ * C_;
  k_transpose_x<<<dim3(T_ / 32, C_ / 32, B_), dim3(32, 8), 0, stream>>>(x, xTh2);
  // temporal conv heads 0,1  (block tile 128x64, 4 waves of 32x64)
  k_tconv<<<dim3(T_ / 64, C_ / 128, TL_ * B_), 128, 0, stream>>>(convW, xTh2, featsT);
  // g/theta/phi projections
  k_proj<<<dim3(T_ / 64, 768 / 128, H_ * B_), 128, 0, stream>>>(projW, projB, featsT, gbuf, thT, phT);
  // flash attention
  k_attn<<<dim3(T_ / 64, H_ * B_), 128, 0, stream>>>(thT, phT, gbuf, yT);
  // W conv + BN1 + residual
  k_wconv<<<dim3(T_ / 64, C_ / 128, B_), 128, 0, stream>>>(Wb16, W_b, bn1g, bn1b, bn1m, bn1v, yT, x, zT);
  // fx conv + BN2 -> out
  k_fx<<<dim3(T_ / 64, OUT_ / 128, B_), 128, 0, stream>>>(fxb16, fx_b, bn2g, bn2b, bn2m, bn2v, zT,
                                                          (float*)d_out);
}
